// ScaledDotProductAttention_relative_7421703487708
// MI455X (gfx1250) — matrix operand order repaired and checked by reading, not compiled, not run on hardware
//
#include <hip/hip_runtime.h>

// ---------------------------------------------------------------------------
// Masked attention for MI455X (gfx1250, wave32, WMMA).
//   q,k,v: [16,2048,64] f32   (k2 unused by reference)
//   mask:  [16,1,2048] bool   edge: [2,2048,2048] i32
//   out = concat( O:[16,2048,64] f32 , attn:[16,2048,2048] f32 )
// One 256-thread block (8 waves) per (head, 16-row tile). Scores for the
// whole 16x2048 strip live in LDS (f32); both GEMMs use
// v_wmma_f32_16x16x32_f16 with f32 accumulation. attn is streamed out with
// non-temporal stores (write-once, keep L2 for edge/K/V).
// ---------------------------------------------------------------------------

typedef __attribute__((ext_vector_type(16))) _Float16 v16h;
typedef __attribute__((ext_vector_type(8)))  float    v8f;

#define BH 16
#define NN 2048
#define DD 64
#define SSTR 2052              // padded LDS row stride (floats): 8*2052 % 64 != 0
#define NEG_INF (-1e15f)

__global__ __launch_bounds__(256)
void attn_edge_kernel(const float* __restrict__ q,
                      const float* __restrict__ k,
                      const float* __restrict__ v,
                      const unsigned char* __restrict__ mask,
                      const int* __restrict__ edge,
                      float* __restrict__ outO,
                      float* __restrict__ outA)
{
    __shared__ float sS[16 * SSTR];   // scores -> probabilities (f32)
    __shared__ float sO[16 * DD];     // cross-wave O reduction

    const int tid  = threadIdx.x;
    const int wave = tid >> 5;
    const int lane = tid & 31;
    const int r16  = lane & 15;       // row (A) or column (B/C) index
    const int half = lane >> 4;       // lane-group select
    const int bh   = blockIdx.y;
    const int n0   = blockIdx.x * 16;

    // ---------------- Q A-fragments (16x64, two K=32 chunks) ----------------
    // A layout: lane holds row r16; VGPR i packs K = half*16 + i*2 (+1)
    const float* qrow = q + ((size_t)bh * NN + n0 + r16) * DD;
    v16h aq0, aq1;
#pragma unroll
    for (int i = 0; i < 8; ++i) {
        const int d0 = (half << 4) + (i << 1);
        float2 p0 = *(const float2*)(qrow + d0);
        float2 p1 = *(const float2*)(qrow + 32 + d0);
        aq0[2 * i]     = (_Float16)p0.x;  aq0[2 * i + 1] = (_Float16)p0.y;
        aq1[2 * i]     = (_Float16)p1.x;  aq1[2 * i + 1] = (_Float16)p1.y;
    }

    const float*         kbh   = k    + (size_t)bh * NN * DD;
    const unsigned char* mrow  = mask + (size_t)bh * NN;
    const int*           ebase = edge + (size_t)(bh & 1) * NN * NN;
    const int            koff  = half << 4;   // B frag: lanes 0-15 -> K0..15, 16-31 -> K16..31
    const int            rbase = half << 3;

    // ---------------- Phase 1: S = QK^T/8, edge+mask, into LDS --------------
#pragma unroll 2
    for (int j = wave; j < NN / 16; j += 8) {
        const int m0  = j * 16;
        const int col = m0 + r16;

        // Issue ALL loads for this tile up front (K frag, mask, 8 edge rows)
        const float* kp = kbh + (size_t)(m0 + r16) * DD + koff;
        float4 kf[8];
#pragma unroll
        for (int t = 0; t < 4; ++t) {
            kf[t]     = *(const float4*)(kp + 4 * t);
            kf[t + 4] = *(const float4*)(kp + 32 + 4 * t);
        }
        const int mvi = (int)mrow[col];                  // per-col mask byte
        const int* ep = ebase + (size_t)(n0 + rbase) * NN + col;
        int ev[8];
#pragma unroll
        for (int i = 0; i < 8; ++i) ev[i] = ep[(size_t)i * NN];

        // Convert K frag to f16
        v16h b0, b1;
#pragma unroll
        for (int t = 0; t < 4; ++t) {
            b0[4*t]   = (_Float16)kf[t].x;   b0[4*t+1] = (_Float16)kf[t].y;
            b0[4*t+2] = (_Float16)kf[t].z;   b0[4*t+3] = (_Float16)kf[t].w;
            b1[4*t]   = (_Float16)kf[t+4].x; b1[4*t+1] = (_Float16)kf[t+4].y;
            b1[4*t+2] = (_Float16)kf[t+4].z; b1[4*t+3] = (_Float16)kf[t+4].w;
        }
        v8f acc = {0.f,0.f,0.f,0.f,0.f,0.f,0.f,0.f};
        acc = __builtin_amdgcn_wmma_f32_16x16x32_f16(false, aq0, false, b0,
                                                     (short)0, acc, false, false);
        acc = __builtin_amdgcn_wmma_f32_16x16x32_f16(false, aq1, false, b1,
                                                     (short)0, acc, false, false);

        // C layout: VGPR i -> row (i + half*8), col (m0 + r16). Branchless mask.
        float* sdst = sS + rbase * SSTR + col;
#pragma unroll
        for (int i = 0; i < 8; ++i) {
            const int  dead = (ev[i] == 0) | mvi;        // bitwise: no branches
            const float s   = dead ? NEG_INF : acc[i] * 0.125f;
            sdst[i * SSTR]  = s;
        }
    }

    for (int e = tid; e < 16 * DD; e += 256) sO[e] = 0.f;   // zero O reduction buf
    __syncthreads();

    // ---------------- Phase 2: row softmax + stream attn out ----------------
    for (int rr = 0; rr < 2; ++rr) {
        const int r = wave * 2 + rr;
        float* srow = sS + r * SSTR;
        float mx = -3.4e38f;
        for (int c = lane; c < NN; c += 32) mx = fmaxf(mx, srow[c]);
#pragma unroll
        for (int off = 16; off; off >>= 1) mx = fmaxf(mx, __shfl_xor(mx, off, 32));
        float sum = 0.f;
        for (int c = lane; c < NN; c += 32) {
            float e = __expf(srow[c] - mx);
            srow[c] = e;
            sum += e;
        }
#pragma unroll
        for (int off = 16; off; off >>= 1) sum += __shfl_xor(sum, off, 32);
        const float inv = 1.f / sum;
        float* arow = outA + ((size_t)bh * NN + n0 + r) * NN;
        for (int c = lane; c < NN; c += 32) {      // 128B-coalesced NT stores
            float p = srow[c] * inv;
            srow[c] = p;
            __builtin_nontemporal_store(p, arow + c);
        }
    }
    __syncthreads();

    // ---------------- Phase 3: O = P·V ----------------
    const float* vbh = v + (size_t)bh * NN * DD;
    v8f o[4];
#pragma unroll
    for (int d = 0; d < 4; ++d) o[d] = (v8f){0.f,0.f,0.f,0.f,0.f,0.f,0.f,0.f};

    for (int ch = wave; ch < NN / 32; ch += 8) {
        const int m0 = ch * 32;
        // A fragment of P from LDS (row r16, K-chunk m0..m0+31)
        const float* prow = sS + r16 * SSTR + m0;
        v16h ap;
#pragma unroll
        for (int i = 0; i < 8; ++i) {
            const int mb = (half << 4) + (i << 1);
            float2 p = *(const float2*)(prow + mb);
            ap[2 * i]     = (_Float16)p.x;
            ap[2 * i + 1] = (_Float16)p.y;
        }
#pragma unroll
        for (int d = 0; d < 4; ++d) {
            // B fragment: lane holds column (d*16 + r16) of V rows m0+koff..+15
            const float* vp = vbh + (size_t)(m0 + koff) * DD + d * 16 + r16;
            v16h bv;
#pragma unroll
            for (int t = 0; t < 16; ++t) bv[t] = (_Float16)vp[(size_t)t * DD];
            o[d] = __builtin_amdgcn_wmma_f32_16x16x32_f16(false, ap, false, bv,
                                                          (short)0, o[d], false, false);
        }
    }
    // cross-wave reduction of partial O via LDS float atomics (ds_add_f32)
#pragma unroll
    for (int d = 0; d < 4; ++d) {
        const int col = d * 16 + r16;
#pragma unroll
        for (int i = 0; i < 8; ++i) {
            const int row = i + (half << 3);
            atomicAdd(&sO[row * DD + col], o[d][i]);
        }
    }
    __syncthreads();

    float* orow = outO + ((size_t)bh * NN + n0) * DD;
    for (int e = tid; e < 16 * DD; e += 256) orow[e] = sO[e];
}

extern "C" void kernel_launch(void* const* d_in, const int* in_sizes, int n_in,
                              void* d_out, int out_size, void* d_ws, size_t ws_size,
                              hipStream_t stream)
{
    const float*         q    = (const float*)d_in[0];
    const float*         k    = (const float*)d_in[1];
    const float*         v    = (const float*)d_in[2];
    // d_in[3] (k2) is unused by the reference computation
    const unsigned char* mask = (const unsigned char*)d_in[4];
    const int*           edge = (const int*)d_in[5];

    float* outO = (float*)d_out;                       // [16,2048,64]
    float* outA = outO + (size_t)BH * NN * DD;         // [16,2048,2048]

    dim3 grid(NN / 16, BH);
    attn_edge_kernel<<<grid, 256, 0, stream>>>(q, k, v, mask, edge, outO, outA);
    (void)in_sizes; (void)n_in; (void)out_size; (void)d_ws; (void)ws_size;
}
